// SMPLModel_29463475651122
// MI455X (gfx1250) — matrix operand order repaired and checked by reading, not compiled, not run on hardware
//
#include <hip/hip_runtime.h>

#define N_JOINTS 24

typedef __attribute__((ext_vector_type(16))) __bf16          v16bf;
typedef __attribute__((ext_vector_type(8)))  float           v8f;
typedef __attribute__((ext_vector_type(8)))  unsigned short  v8u;
typedef __attribute__((ext_vector_type(16))) unsigned short  v16u;

__device__ __forceinline__ unsigned short f32_to_bf16(float f) {
    union { float f; unsigned int u; } x; x.f = f;
    unsigned int u = x.u;
    u += 0x7fffu + ((u >> 16) & 1u);          // round-to-nearest-even
    return (unsigned short)(u >> 16);
}

// ---------------------------------------------------------------------------
// Kernel 1: per-batch Rodrigues + forward kinematics + rest-pose correction.
// Emits Bmat[n=3b+x][k=j*4+y] = G_corrected[b,j,x,y] as bf16 (K-major rows,
// 96 elements = 192 B per row -> exactly what the GEMM B-fragment loads want).
// ---------------------------------------------------------------------------
__global__ void smpl_build_G(const float* __restrict__ J_hat,
                             const float* __restrict__ pose,
                             unsigned short* __restrict__ Bpack, int B) {
    int b = blockIdx.x * blockDim.x + threadIdx.x;
    if (b >= B) return;

    const int par[N_JOINTS] = {-1,0,0,0,1,2,3,4,5,6,7,8,9,9,9,
                               12,13,14,16,17,18,19,20,21};
    float Gr[N_JOINTS][9];
    float Gt[N_JOINTS][3];

    #pragma unroll
    for (int j = 0; j < N_JOINTS; ++j) {
        float rx = pose[b * 72 + 3 * j + 0];
        float ry = pose[b * 72 + 3 * j + 1];
        float rz = pose[b * 72 + 3 * j + 2];
        float th  = sqrtf(rx * rx + ry * ry + rz * rz) + 1e-8f;
        float inv = 1.0f / th;
        float ux = rx * inv, uy = ry * inv, uz = rz * inv;
        float c = cosf(th), s = sinf(th), ic = 1.0f - c;
        float R[9];
        R[0] = c + ic * ux * ux;     R[1] = ic * ux * uy - s * uz; R[2] = ic * ux * uz + s * uy;
        R[3] = ic * uy * ux + s * uz; R[4] = c + ic * uy * uy;     R[5] = ic * uy * uz - s * ux;
        R[6] = ic * uz * ux - s * uy; R[7] = ic * uz * uy + s * ux; R[8] = c + ic * uz * uz;

        if (j == 0) {
            #pragma unroll
            for (int t = 0; t < 9; ++t) Gr[0][t] = R[t];
            Gt[0][0] = J_hat[0]; Gt[0][1] = J_hat[1]; Gt[0][2] = J_hat[2];
        } else {
            const int p = par[j];
            float relx = J_hat[3 * j + 0] - J_hat[3 * p + 0];
            float rely = J_hat[3 * j + 1] - J_hat[3 * p + 1];
            float relz = J_hat[3 * j + 2] - J_hat[3 * p + 2];
            #pragma unroll
            for (int x = 0; x < 3; ++x) {
                #pragma unroll
                for (int y = 0; y < 3; ++y)
                    Gr[j][3 * x + y] = Gr[p][3 * x + 0] * R[0 + y]
                                     + Gr[p][3 * x + 1] * R[3 + y]
                                     + Gr[p][3 * x + 2] * R[6 + y];
                Gt[j][x] = Gr[p][3 * x + 0] * relx + Gr[p][3 * x + 1] * rely
                         + Gr[p][3 * x + 2] * relz + Gt[p][x];
            }
        }
    }

    #pragma unroll
    for (int x = 0; x < 3; ++x) {
        unsigned short* row = Bpack + (size_t)(3 * b + x) * 96;
        #pragma unroll
        for (int j = 0; j < N_JOINTS; ++j) {
            float jx = J_hat[3 * j + 0], jy = J_hat[3 * j + 1], jz = J_hat[3 * j + 2];
            float tc = Gt[j][x] - (Gr[j][3 * x + 0] * jx + Gr[j][3 * x + 1] * jy
                                 + Gr[j][3 * x + 2] * jz);
            row[4 * j + 0] = f32_to_bf16(Gr[j][3 * x + 0]);
            row[4 * j + 1] = f32_to_bf16(Gr[j][3 * x + 1]);
            row[4 * j + 2] = f32_to_bf16(Gr[j][3 * x + 2]);
            row[4 * j + 3] = f32_to_bf16(tc);
        }
    }
}

// ---------------------------------------------------------------------------
// Kernel 2: A[v][j*4+y] = weights[v,j] * [T_hat[v], 1][y], bf16, row-major,
// rows padded with zeros up to Vpad (multiple of 16).
// ---------------------------------------------------------------------------
__global__ void smpl_build_A(const float* __restrict__ T_hat,
                             const float* __restrict__ W,
                             unsigned short* __restrict__ Abf,
                             int V, int Vpad) {
    int v = blockIdx.x * blockDim.x + threadIdx.x;
    if (v >= Vpad) return;
    unsigned int* row = (unsigned int*)(Abf + (size_t)v * 96);
    if (v >= V) {
        #pragma unroll 4
        for (int i = 0; i < 48; ++i) row[i] = 0u;
        return;
    }
    float tx = T_hat[3 * v + 0], ty = T_hat[3 * v + 1], tz = T_hat[3 * v + 2];
    #pragma unroll
    for (int j = 0; j < N_JOINTS; ++j) {
        float w = W[v * N_JOINTS + j];
        unsigned int lo = (unsigned int)f32_to_bf16(w * tx)
                        | ((unsigned int)f32_to_bf16(w * ty) << 16);
        unsigned int hh = (unsigned int)f32_to_bf16(w * tz)
                        | ((unsigned int)f32_to_bf16(w) << 16);
        row[2 * j + 0] = lo;
        row[2 * j + 1] = hh;
    }
}

// ---------------------------------------------------------------------------
// Kernel 3: the GEMM  out(V x 3B) = A(V x 96) * B^T(96 x 3B)  via bf16 WMMA.
// One wave computes a 16(M) x 48(N) strip = 16 vertices x 16 full batches,
// 9 x v_wmma_f32_16x16x32_bf16, staged through LDS for coalesced NT stores.
// ---------------------------------------------------------------------------
__global__ __launch_bounds__(256) void smpl_lbs_wmma(
    const unsigned short* __restrict__ Abf,
    const unsigned short* __restrict__ Bpack,
    const float* __restrict__ trans,
    float* __restrict__ out,
    int V, int B, int Mtiles, int totalWaves)
{
    __shared__ float lds[8][16 * 52];     // 52-float row stride: conflict-free
    const int wib = threadIdx.x >> 5;
    const int wid = blockIdx.x * 8 + wib;
    if (wid >= totalWaves) return;        // wave-uniform: EXEC stays all-ones

    const int mtile  = wid % Mtiles;
    const int nstrip = wid / Mtiles;
    const int m0 = mtile * 16;
    const int n0 = nstrip * 48;
    const int lane = threadIdx.x & 31;
    const int lof  = lane & 15;
    const int hi   = lane >> 4;

    const unsigned short* aRow = Abf + (size_t)(m0 + lof) * 96;

    v8f acc[3];
    #pragma unroll
    for (int s = 0; s < 3; ++s)
        acc[s] = (v8f){0.f, 0.f, 0.f, 0.f, 0.f, 0.f, 0.f, 0.f};

    #pragma unroll
    for (int kk = 0; kk < 3; ++kk) {
        const int k0 = kk * 32;
        // A fragment, contiguous K order (matches B):
        //   lanes 0-15 : K = k0+0..15   (two b128 loads)
        //   lanes 16-31: K = k0+16..31
        v8u alo = *(const v8u*)(aRow + k0 + hi * 16);
        v8u ahi = *(const v8u*)(aRow + k0 + hi * 16 + 8);
        v16u au = __builtin_shufflevector(alo, ahi,
                   0, 1, 2, 3, 4, 5, 6, 7, 8, 9, 10, 11, 12, 13, 14, 15);
        v16bf a = __builtin_bit_cast(v16bf, au);
        #pragma unroll
        for (int sub = 0; sub < 3; ++sub) {
            // B fragment: lane holds one N-column, K k0+hi*16 .. +15 (32 B)
            const unsigned short* bCol =
                Bpack + (size_t)(n0 + sub * 16 + lof) * 96 + k0 + hi * 16;
            v16bf bf = __builtin_bit_cast(v16bf, *(const v16u*)bCol);
            acc[sub] = __builtin_amdgcn_wmma_f32_16x16x32_bf16(
                false, a, false, bf, (short)0, acc[sub], false, false);
        }
    }

    // Stage (acc + trans) into LDS: lds[m_local][n_local]
    float* sl = &lds[wib][0];
    #pragma unroll
    for (int sub = 0; sub < 3; ++sub) {
        float t = trans[n0 + sub * 16 + lof];    // trans flat index == n
        #pragma unroll
        for (int r = 0; r < 8; ++r)
            sl[(r + 8 * hi) * 52 + sub * 16 + lof] = acc[sub][r] + t;
    }
    asm volatile("s_wait_dscnt 0" ::: "memory");

    // Coalesced write-out: for each batch bb the 16 vertices form a
    // contiguous 192-byte segment at out + (b0+bb)*3V + m0*3.
    const int b0 = nstrip * 16;
    const size_t V3 = (size_t)3 * V;
    #pragma unroll
    for (int it = 0; it < 24; ++it) {
        int flat = it * 32 + lane;               // 0..767 = 16 batches * 48
        int bb = flat / 48;
        int i  = flat - bb * 48;                 // m_local*3 + x
        int ml = i / 3;
        int x  = i - ml * 3;
        if ((m0 + ml < V) && (b0 + bb < B)) {
            float val = sl[ml * 52 + bb * 3 + x];
            __builtin_nontemporal_store(
                val, out + (size_t)(b0 + bb) * V3 + (size_t)m0 * 3 + i);
        }
    }
}

// ---------------------------------------------------------------------------
extern "C" void kernel_launch(void* const* d_in, const int* in_sizes, int n_in,
                              void* d_out, int out_size, void* d_ws, size_t ws_size,
                              hipStream_t stream) {
    const float* T_hat = (const float*)d_in[0];
    const float* J_hat = (const float*)d_in[1];
    const float* W     = (const float*)d_in[2];
    const float* pose  = (const float*)d_in[3];
    const float* trans = (const float*)d_in[4];
    float* out = (float*)d_out;

    const int V = in_sizes[0] / 3;          // 6890
    const int B = in_sizes[3] / 72;         // 512
    const int Mtiles  = (V + 15) / 16;      // 431
    const int Vpad    = Mtiles * 16;        // 6896
    const int Nstrips = (B + 15) / 16;      // 32

    unsigned short* Abf = (unsigned short*)d_ws;
    size_t abfBytes = ((size_t)Vpad * 96 * sizeof(unsigned short) + 255) & ~(size_t)255;
    unsigned short* Bpack = (unsigned short*)((char*)d_ws + abfBytes);

    smpl_build_G<<<(B + 127) / 128, 128, 0, stream>>>(J_hat, pose, Bpack, B);
    smpl_build_A<<<(Vpad + 255) / 256, 256, 0, stream>>>(T_hat, W, Abf, V, Vpad);

    const int totalWaves = Mtiles * Nstrips;          // 13792
    const int blocks = (totalWaves + 7) / 8;          // 1724 (exact fit)
    smpl_lbs_wmma<<<blocks, 256, 0, stream>>>(Abf, Bpack, trans, out,
                                              V, B, Mtiles, totalWaves);
}
